// Attention_833223655434
// MI455X (gfx1250) — compile-verified, matrix operand order checked by reading, not run on hardware
//
#include <hip/hip_runtime.h>

// ---------------------------------------------------------------------------
// MI455X (gfx1250) attention pipeline: bf16 WMMA GEMMs + flash attention.
// wave32; v_wmma_f32_16x16x32_bf16 everywhere; f32 accumulation.
// GEMMs: 128x128 block tiles, double-buffered k=32 LDS panels staged with
// GLOBAL_LOAD_ASYNC_TO_LDS_B128 (ASYNCcnt) so the DMA overlaps the WMMAs.
// ---------------------------------------------------------------------------

#define DEV __device__ __forceinline__

typedef __attribute__((ext_vector_type(16))) __bf16     bf16x16;
typedef __attribute__((ext_vector_type(8)))  float       f32x8;
typedef __attribute__((ext_vector_type(4)))  unsigned int uint4v;

union FragU { bf16x16 bf; uint4v u[2]; };

static constexpr int NB = 2;          // batch
static constexpr int T  = 2048;       // seq len
static constexpr int C  = 1024;       // embed
static constexpr int H  = 16;         // heads
static constexpr int D  = 64;         // head dim
static constexpr int MR = NB * T;     // 4096 rows of x
static constexpr int PAD = 40;        // LDS row stride (elems); 80B keeps the
                                      // 16-lane fragment reads on distinct banks

// ---- WMMA helpers ---------------------------------------------------------

DEV f32x8 wmma_bf16(bf16x16 a, bf16x16 b, f32x8 c) {
  return __builtin_amdgcn_wmma_f32_16x16x32_bf16(
      /*neg_a=*/false, a, /*neg_b=*/false, b,
      /*c_mod=*/(short)0, c, /*reuse_a=*/false, /*reuse_b=*/false);
}

// Load a 16x32 bf16 fragment (A layout; B identical with N<->M) from a
// row-major [16][ld] tile.  ISA layout: lanes 0-15 hold K=0..7 (u[0]) and
// K=16..23 (u[1]); lanes 16-31 hold K=8..15 / K=24..31.  Two b128 per lane.
DEV bf16x16 load_frag(const __bf16* base, int ld) {
  const int lane = threadIdx.x & 31;
  const int r    = lane & 15;
  const int hi   = lane >> 4;
  const __bf16* p = base + (size_t)r * ld + hi * 8;
  FragU f;
  f.u[0] = *(const uint4v*)(p);
  f.u[1] = *(const uint4v*)(p + 16);
  return f.bf;
}

// Async 16B global -> LDS copy (CDNA5 GLOBAL_LOAD_ASYNC_TO_LDS_B128,
// tracked by ASYNCcnt).  ldsOff is the LDS byte address (generic shared
// pointers map addr[31:0] onto the LDS aperture).
DEV void async_ld16(unsigned ldsOff, const __bf16* g) {
  asm volatile("global_load_async_to_lds_b128 %0, %1, off"
               :: "v"(ldsOff), "v"(g) : "memory");
}
DEV void wait_async0() {
  asm volatile("s_wait_asynccnt 0" ::: "memory");
}

// ---- Kernel 1: f32 -> bf16 ------------------------------------------------

__global__ __launch_bounds__(256) void cvt_bf16_kernel(
    const float* __restrict__ in, __bf16* __restrict__ out, int n) {
  int i = blockIdx.x * blockDim.x + threadIdx.x;
  if (i < n) out[i] = (__bf16)in[i];
}

// ---- Kernel 2/5: GEMM  out[M,N] = A[M,K] * W[N,K]^T -----------------------
// 256 threads = 8 waves.  Block tile 128x128; wave tile 32x64 (2x4 WMMA
// tiles).  K consumed in chunks of 32 through double-buffered LDS panels
// filled by async DMA that overlaps the previous chunk's WMMAs.

template <bool BF16OUT>
__global__ __launch_bounds__(256) void gemm_lds_kernel(
    const __bf16* __restrict__ A, const __bf16* __restrict__ W,
    void* __restrict__ out, int M, int N, int K) {
  __shared__ __attribute__((aligned(16))) __bf16 ldsA[2][128 * PAD];
  __shared__ __attribute__((aligned(16))) __bf16 ldsB[2][128 * PAD];

  const int nblk = N >> 7;
  const int m0 = (blockIdx.x / nblk) << 7;
  const int n0 = (blockIdx.x % nblk) << 7;

  const int tid  = threadIdx.x;
  const int wave = tid >> 5;
  const int lane = tid & 31;
  const int wm = (wave & 3) * 32;   // wave M offset in block tile
  const int wn = (wave >> 2) * 64;  // wave N offset in block tile

  // Staging: each thread DMAs 2x16B of A and 2x16B of B per k-step.
  const int srow   = tid >> 2;        // 0..63
  const int schunk = (tid & 3) * 8;   // elem offset 0,8,16,24

  auto stage_async = [&](int k0, int buf) {
    async_ld16((unsigned)(uintptr_t)&ldsA[buf][(srow)      * PAD + schunk],
               A + (size_t)(m0 + srow)      * K + k0 + schunk);
    async_ld16((unsigned)(uintptr_t)&ldsA[buf][(srow + 64) * PAD + schunk],
               A + (size_t)(m0 + srow + 64) * K + k0 + schunk);
    async_ld16((unsigned)(uintptr_t)&ldsB[buf][(srow)      * PAD + schunk],
               W + (size_t)(n0 + srow)      * K + k0 + schunk);
    async_ld16((unsigned)(uintptr_t)&ldsB[buf][(srow + 64) * PAD + schunk],
               W + (size_t)(n0 + srow + 64) * K + k0 + schunk);
  };

  f32x8 acc[2][4] = {};
  stage_async(0, 0);
  wait_async0();
  __syncthreads();

  const int NK = K >> 5;
  for (int ks = 0; ks < NK; ++ks) {
    const int buf = ks & 1;
    // Prefetch next panel into the other buffer; DMA overlaps the WMMAs.
    if (ks + 1 < NK) stage_async((ks + 1) << 5, buf ^ 1);

    bf16x16 af0 = load_frag(&ldsA[buf][(wm +  0) * PAD], PAD);
    bf16x16 af1 = load_frag(&ldsA[buf][(wm + 16) * PAD], PAD);
#pragma unroll
    for (int j = 0; j < 4; ++j) {
      bf16x16 bf = load_frag(&ldsB[buf][(wn + j * 16) * PAD], PAD);
      acc[0][j] = wmma_bf16(af0, bf, acc[0][j]);
      acc[1][j] = wmma_bf16(af1, bf, acc[1][j]);
    }

    wait_async0();       // this wave's DMA into buf^1 has landed
    __syncthreads();     // everyone's DMA has landed; buf free for reuse
  }

  // C layout: lane (0-15) holds col, rows +0..7 in acc[.][j][0..7];
  // lanes 16-31 hold rows +8..15.
  const int col = lane & 15;
  const int rb  = (lane >> 4) * 8;
#pragma unroll
  for (int mi = 0; mi < 2; ++mi) {
#pragma unroll
    for (int j = 0; j < 4; ++j) {
#pragma unroll
      for (int i = 0; i < 8; ++i) {
        size_t idx = (size_t)(m0 + wm + mi * 16 + rb + i) * N +
                     (n0 + wn + j * 16 + col);
        if (BF16OUT) ((__bf16*)out)[idx] = (__bf16)acc[mi][j][i];
        else         ((float*)out)[idx]  = acc[mi][j][i];
      }
    }
  }
}

// ---- Kernel 3: RoPE + head split + V transpose ----------------------------

__global__ __launch_bounds__(256) void rope_kernel(
    const __bf16* __restrict__ qkv, __bf16* __restrict__ Qo,
    __bf16* __restrict__ Ko, __bf16* __restrict__ Vt) {
  int idx = blockIdx.x * blockDim.x + threadIdx.x;   // ((b*H+h)*T+t)*D+d
  int d  = idx & 63;
  int tt = (idx >> 6) & (T - 1);
  int h  = (idx >> 17) & (H - 1);
  int b  = idx >> 21;

  size_t row = (size_t)(b * T + tt) * (3 * C) + h * D;
  float qv = (float)qkv[row + d];
  float qp = (float)qkv[row + (d ^ 32)];
  float kv = (float)qkv[row + C + d];
  float kp = (float)qkv[row + C + (d ^ 32)];
  float vv = (float)qkv[row + 2 * C + d];

  float ang = (float)tt * __expf(-__logf(10000.0f) * (float)(d & 31) * (1.0f / 32.0f));
  float cs = __cosf(ang), sn = __sinf(ang);
  float sign = (d < 32) ? -1.0f : 1.0f;

  Qo[idx] = (__bf16)((qv * cs + sign * qp * sn) * 0.125f);  // fold 1/sqrt(64)
  Ko[idx] = (__bf16)(kv * cs + sign * kp * sn);
  Vt[((size_t)(b * H + h) * D + d) * T + tt] = (__bf16)vv;
}

// ---- Kernel 4: flash attention -------------------------------------------
// One wave owns a 16-row q tile of one (b,h).  Per 32-key step: 4 WMMAs for
// S, row-max via 16-lane xor butterfly, P through per-wave LDS (C->A layout
// change, s_wait_dscnt fence), row-sum via one P x ones WMMA, 4 WMMAs for PV.

__global__ __launch_bounds__(256) void attn_kernel(
    const __bf16* __restrict__ Q, const __bf16* __restrict__ Kr,
    const __bf16* __restrict__ Vt, __bf16* __restrict__ Y) {
  __shared__ __attribute__((aligned(16))) __bf16 smem[8 * 16 * 32];

  const int wib  = threadIdx.x >> 5;
  const int wave = blockIdx.x * 8 + wib;
  const int bh   = wave >> 7;              // /(T/16)
  const int qt   = wave & 127;
  const int b    = bh >> 4;
  const int h    = bh & 15;

  const int lane = threadIdx.x & 31;
  const int col  = lane & 15;
  const int rb   = (lane >> 4) * 8;
  __bf16* lds = &smem[wib * 512];

  const __bf16* qbase = Q  + (size_t)bh * T * D + (size_t)(qt * 16) * D;
  const __bf16* kbase = Kr + (size_t)bh * T * D;
  const __bf16* vbase = Vt + (size_t)bh * D * T;

  bf16x16 qa0 = load_frag(qbase, D);
  bf16x16 qa1 = load_frag(qbase + 32, D);

  FragU onesf;                              // B = all-ones 32x16 (bf16 1.0)
#pragma unroll
  for (int i = 0; i < 4; ++i) { onesf.u[0][i] = 0x3F803F80u; onesf.u[1][i] = 0x3F803F80u; }

  f32x8 o[4] = {};
  float m[8], l[8], corr_a[8];
#pragma unroll
  for (int i = 0; i < 8; ++i) { m[i] = -1e30f; l[i] = 0.0f; }

  const int qhi = qt * 16 + 15;
  for (int kj = 0; kj <= qhi; kj += 32) {
    // S tiles: rows = q, cols = keys kj..kj+31
    f32x8 s0 = {}, s1 = {};
    {
      const __bf16* kp0 = kbase + (size_t)kj * D;
      const __bf16* kp1 = kbase + (size_t)(kj + 16) * D;
      s0 = wmma_bf16(qa0, load_frag(kp0, D), s0);
      s0 = wmma_bf16(qa1, load_frag(kp0 + 32, D), s0);
      s1 = wmma_bf16(qa0, load_frag(kp1, D), s1);
      s1 = wmma_bf16(qa1, load_frag(kp1 + 32, D), s1);
    }

    // Causal mask + row max (xor butterfly over the 16 lanes of a half).
#pragma unroll
    for (int i = 0; i < 8; ++i) {
      int qr = qt * 16 + rb + i;
      float e0 = (kj + col      > qr) ? -1e30f : s0[i];
      float e1 = (kj + 16 + col > qr) ? -1e30f : s1[i];

      float rmax = fmaxf(e0, e1);
#pragma unroll
      for (int msk = 1; msk < 16; msk <<= 1)
        rmax = fmaxf(rmax, __shfl_xor(rmax, msk, 32));

      float mn   = fmaxf(m[i], rmax);
      float corr = __expf(m[i] - mn);
      corr_a[i] = corr;
      m[i] = mn;

      lds[(rb + i) * 32 + col]      = (__bf16)__expf(e0 - mn);
      lds[(rb + i) * 32 + 16 + col] = (__bf16)__expf(e1 - mn);
#pragma unroll
      for (int j = 0; j < 4; ++j) o[j][i] *= corr;
    }
    asm volatile("s_wait_dscnt 0" ::: "memory");   // same-wave DS RAW fence

    // Re-read P as an A fragment (16x32 row-major in LDS).
    FragU pf;
    {
      const __bf16* pp = lds + (lane & 15) * 32 + (lane >> 4) * 8;
      pf.u[0] = *(const uint4v*)(pp);
      pf.u[1] = *(const uint4v*)(pp + 16);
    }

    // Row sums from the matrix core: P x ones -> every lane gets rowsum(rb+i).
    f32x8 zero = {};
    f32x8 rsv = wmma_bf16(pf.bf, onesf.bf, zero);
#pragma unroll
    for (int i = 0; i < 8; ++i) l[i] = l[i] * corr_a[i] + rsv[i];

    // O += P * V  (B[k][n] = Vt[n][k], K-contiguous)
#pragma unroll
    for (int j = 0; j < 4; ++j) {
      bf16x16 vb = load_frag(vbase + (size_t)(j * 16) * T + kj, T);
      o[j] = wmma_bf16(pf.bf, vb, o[j]);
    }
  }

  // Normalize; write y as [b][t][h*D+d] bf16 for the out-proj GEMM.
#pragma unroll
  for (int j = 0; j < 4; ++j) {
#pragma unroll
    for (int i = 0; i < 8; ++i) {
      int tt = qt * 16 + rb + i;
      float val = o[j][i] / l[i];
      Y[(size_t)(b * T + tt) * C + h * D + j * 16 + col] = (__bf16)val;
    }
  }
}

// ---- Host launch ----------------------------------------------------------

extern "C" void kernel_launch(void* const* d_in, const int* in_sizes, int n_in,
                              void* d_out, int out_size, void* d_ws, size_t ws_size,
                              hipStream_t stream) {
  (void)in_sizes; (void)n_in; (void)out_size; (void)ws_size;
  const float* x      = (const float*)d_in[0];
  const float* w_qkv  = (const float*)d_in[1];
  const float* w_proj = (const float*)d_in[2];
  float* out = (float*)d_out;

  const int NX   = NB * T * C;        // 4194304
  const int NWQ  = 3 * C * C;         // 3145728
  const int NWP  = C * C;             // 1048576
  const int NQKV = NB * T * 3 * C;    // 12582912

  size_t off = 0;
  auto carve = [&](size_t elems) {
    char* p = (char*)d_ws + off;
    off += (elems * sizeof(__bf16) + 255) & ~(size_t)255;
    return (__bf16*)p;
  };
  __bf16* xb     = carve(NX);
  __bf16* wqkvb  = carve(NWQ);
  __bf16* wprojb = carve(NWP);
  __bf16* qkvb   = carve(NQKV);
  __bf16* Qr     = carve(NX);
  __bf16* Kb     = carve(NX);
  __bf16* Vt     = carve(NX);
  __bf16* Yb     = carve(NX);

  // 1) convert inputs to bf16
  cvt_bf16_kernel<<<(NX  + 255) / 256, 256, 0, stream>>>(x,      xb,     NX);
  cvt_bf16_kernel<<<(NWQ + 255) / 256, 256, 0, stream>>>(w_qkv,  wqkvb,  NWQ);
  cvt_bf16_kernel<<<(NWP + 255) / 256, 256, 0, stream>>>(w_proj, wprojb, NWP);

  // 2) qkv = x @ w_qkv^T   (M=4096, N=3072, K=1024); 128x128 block tiles
  gemm_lds_kernel<true><<<(MR / 128) * (3 * C / 128), 256, 0, stream>>>(
      xb, wqkvb, qkvb, MR, 3 * C, C);

  // 3) RoPE + head split + V transpose
  rope_kernel<<<NX / 256, 256, 0, stream>>>(qkvb, Qr, Kb, Vt);

  // 4) causal flash attention  (B*H*(T/16) = 4096 waves)
  attn_kernel<<<NB * H * (T / 16) / 8, 256, 0, stream>>>(Qr, Kb, Vt, Yb);

  // 5) out = y @ w_proj^T   (M=4096, N=1024, K=1024), f32 output
  gemm_lds_kernel<false><<<(MR / 128) * (C / 128), 256, 0, stream>>>(
      Yb, wprojb, out, MR, C, C);
}
